// ModulationIndex_11622181503726
// MI455X (gfx1250) — matrix operand order repaired and checked by reading, not compiled, not run on hardware
//
#include <hip/hip_runtime.h>
#include <hip/hip_bf16.h>

// ---------------------------------------------------------------------------
// Modulation Index via WMMA on CDNA5 (gfx1250, wave32).
//
// amp_sums[b,c,p,a,s,k] = sum_t onehot(bins[b,c,p,s,t])[k] * amp[b,c,a,s,t]
// is a GEMM per (b,c,p,s): D[a,k] = A(a,t) x B(t,k), with A = amp in f16 and
// B = one-hot phase mask generated IN REGISTERS from u8 bins in LDS.
// Row a==50 of A is all-ones so D row 50 yields the bin counts for free.
// f32 accumulation via v_wmma_f32_16x16x32_f16.
//
// Decomposition: 320 threads = 10 wave32 waves; wave = (nTile 0..1, pi 0..4).
// Each wave builds its one-hot B-fragment ONCE per K-step (no redundant VALU)
// and reuses it for all 4 M-tiles (a-rows 0..63) -> 40 WMMA tiles per K-step
// per workgroup with zero duplicate mask construction.
// ---------------------------------------------------------------------------

typedef __attribute__((ext_vector_type(16))) _Float16 v16h;
typedef __attribute__((ext_vector_type(8)))  _Float16 v8h;
typedef __attribute__((ext_vector_type(8)))  float    v8f;

#define B_DIM   2
#define C_DIM   16
#define FP_DIM  30
#define FA_DIM  50
#define S_DIM   20
#define T_DIM   1024
#define NB      18          // phase bins
#define PGROUP  5           // p channels per workgroup (30 = 6 groups of 5)
#define NPG     (FP_DIM / PGROUP)
#define NTHREADS 320        // 10 waves = 2 nTiles x 5 p
#define CHUNK   256         // K-chunk (t) staged in LDS per iteration
#define AMP_STRIDE (CHUNK + 8)   // +8 halfs: break 256B LDS bank period
#define SUM_STRIDE 33
#define LOG_NB  2.8903717578961645f   // log(18)
#define PI_F    3.14159265358979323846f
#define BIN_SCALE (18.0f / (2.0f * PI_F))

__global__ __launch_bounds__(NTHREADS)
void ModulationIndex_11622181503726_kernel(const float* __restrict__ pha,
                                           const float* __restrict__ amp,
                                           const float* __restrict__ cuts,
                                           float* __restrict__ out) {
    __shared__ __align__(16) unsigned char binsLds[PGROUP][T_DIM]; // 5 KB
    __shared__ __align__(16) _Float16 ampLds[64][AMP_STRIDE];      // 33 KB
    __shared__ __align__(16) float sumsLds[64][SUM_STRIDE];        // 8.25 KB
    __shared__ float cutLds[NB + 1];

    const int tid  = threadIdx.x;
    const int lane = tid & 31;          // wave32
    const int hl   = lane >> 4;         // half-wave (0/1)
    const int lr   = lane & 15;
    const int wave = tid >> 5;          // 10 waves
    const int nTile = wave & 1;         // 0..1 -> cols k = 16*nTile..+15
    const int piW   = wave >> 1;        // 0..4 -> which p this wave owns
    const int kcol  = 16 * nTile + lr;  // lane's bin column (>=18 -> zeros)

    const int bcpg = blockIdx.x;
    const int pg = bcpg % NPG;
    const int c  = (bcpg / NPG) % C_DIM;
    const int b  =  bcpg / (NPG * C_DIM);
    const int p0 = pg * PGROUP;
    const long bc = (long)b * C_DIM + c;

    if (tid < NB + 1) cutLds[tid] = cuts[tid];

    const v8f vzero = {};
    v8f acc[4];                          // 4 M-tiles (a = 0..63) for (nTile,piW)
#pragma unroll
    for (int m = 0; m < 4; ++m) acc[m] = vzero;

    float miAcc[PGROUP] = {0.f, 0.f, 0.f, 0.f, 0.f};

    for (int s = 0; s < S_DIM; ++s) {
        __syncthreads();   // binsLds/cutLds safe to (re)use

        // ---- phase -> bin: arithmetic guess + exact fix-up vs cutoffs -----
        // searchsorted(cuts, x, 'left')-1 clipped to [0,17]. cnt-1 >= g iff
        // cuts[g] < x; the linspace guess is within +-1 of the true bin.
        const long phaBase = ((bc * FP_DIM + p0) * S_DIM + s) * T_DIM;
        for (int idx = tid; idx < PGROUP * T_DIM; idx += NTHREADS) {
            const int pi = idx >> 10;
            const int t  = idx & (T_DIM - 1);
            const float x = pha[phaBase + (long)pi * (S_DIM * T_DIM) + t];
            int g = (int)floorf((x + PI_F) * BIN_SCALE);
            g = g < 0 ? 0 : (g > NB - 1 ? NB - 1 : g);
            if (cutLds[g + 1] < x)      g += 1;   // true bin above guess
            else if (!(cutLds[g] < x))  g -= 1;   // true bin below guess
            g = g < 0 ? 0 : (g > NB - 1 ? NB - 1 : g);
            binsLds[pi][t] = (unsigned char)g;
        }

        // ---- K loop over t in LDS-staged chunks ---------------------------
        for (int tc = 0; tc < T_DIM / CHUNK; ++tc) {
            const int tbase = tc * CHUNK;
            __syncthreads();   // prior chunk consumed (and bins staged)

            // stage amp chunk as f16, rows: 0..49 amp, 50 = ones, 51..63 = 0
            for (int idx = tid; idx < 64 * (CHUNK / 2); idx += NTHREADS) {
                const int r  = idx >> 7;
                const int t  = (idx & 127) * 2;
                _Float16 h0, h1;
                if (r < FA_DIM) {
                    const float2 v = *(const float2*)
                        &amp[((bc * FA_DIM + r) * S_DIM + s) * T_DIM + tbase + t];
                    h0 = (_Float16)v.x; h1 = (_Float16)v.y;
                } else if (r == FA_DIM) {
                    h0 = (_Float16)1.0f; h1 = (_Float16)1.0f;
                } else {
                    h0 = (_Float16)0.0f; h1 = (_Float16)0.0f;
                }
                union { _Float16 h[2]; unsigned int u; } pk;
                pk.h[0] = h0; pk.h[1] = h1;
                *(unsigned int*)&ampLds[r][t] = pk.u;
            }
            __syncthreads();

            for (int kb = 0; kb < CHUNK; kb += 32) {
                // B fragment (32x16 f16) built ONCE per wave per K-step:
                // lane = column k, K halfs 16*hl..+16; val = (bins[t]==kcol).
                const int tglob = tbase + kb + 16 * hl;   // 16B aligned
                const uint4 w = *(const uint4*)&binsLds[piW][tglob];
                const unsigned int wd[4] = {w.x, w.y, w.z, w.w};
                v16h bfrag;
#pragma unroll
                for (int d = 0; d < 4; ++d)
#pragma unroll
                    for (int j = 0; j < 4; ++j)
                        bfrag[4 * d + j] =
                            ((int)((wd[d] >> (8 * j)) & 0xffu) == kcol)
                                ? (_Float16)1.0f : (_Float16)0.0f;

                // reuse bfrag across all 4 M-tiles (A re-read from LDS)
#pragma unroll
                for (int m = 0; m < 4; ++m) {
                    // A fragment (16x32 f16): lane = row a = 16m+lr, halfs
                    // K = 16*hl..+16 (contiguous, matches B)
                    const _Float16* asrc = &ampLds[16 * m + lr][kb + 16 * hl];
                    const v8h alo = *(const v8h*)asrc;
                    const v8h ahi = *(const v8h*)(asrc + 8);
                    const v16h afrag = __builtin_shufflevector(
                        alo, ahi, 0,1,2,3,4,5,6,7,8,9,10,11,12,13,14,15);

                    acc[m] = __builtin_amdgcn_wmma_f32_16x16x32_f16(
                        false, afrag, false, bfrag, (short)0, acc[m],
                        false, false);
                }
            }
        }

        // ---- epilogue per p: means -> probs -> entropy -> MI --------------
#pragma unroll
        for (int piL = 0; piL < PGROUP; ++piL) {
            __syncthreads();   // sumsLds free
            if (piW == piL) {  // the two waves (nTile 0/1) owning this p store
#pragma unroll
                for (int m = 0; m < 4; ++m) {
#pragma unroll
                    for (int v = 0; v < 8; ++v)  // D: M = v+8*hl, N = lr
                        sumsLds[16 * m + 8 * hl + v][16 * nTile + lr] =
                            acc[m][v];
                    acc[m] = vzero;              // reset for next s
                }
            }
            __syncthreads();

            if (tid < FA_DIM) {
                float mean[NB];
                float tot = 0.f;
#pragma unroll
                for (int k = 0; k < NB; ++k) {
                    const float cnt = sumsLds[FA_DIM][k];   // ones row
                    const float m = sumsLds[tid][k] / (cnt + 1e-9f);
                    mean[k] = m; tot += m;
                }
                const float inv = 1.0f / (tot + 1e-9f);
                float ent = 0.f;
#pragma unroll
                for (int k = 0; k < NB; ++k) {
                    const float pr = mean[k] * inv;
                    ent += pr * __logf(pr + 1e-9f);
                }
                miAcc[piL] += (LOG_NB + ent) / LOG_NB;
            }
        }
    }

    if (tid < FA_DIM) {
#pragma unroll
        for (int pi = 0; pi < PGROUP; ++pi)
            out[(bc * FP_DIM + p0 + pi) * FA_DIM + tid] =
                miAcc[pi] * (1.0f / S_DIM);
    }
}

extern "C" void kernel_launch(void* const* d_in, const int* in_sizes, int n_in,
                              void* d_out, int out_size, void* d_ws, size_t ws_size,
                              hipStream_t stream) {
    (void)in_sizes; (void)n_in; (void)d_ws; (void)ws_size; (void)out_size;
    const float* pha  = (const float*)d_in[0];
    const float* amp  = (const float*)d_in[1];
    const float* cuts = (const float*)d_in[2];
    float* out = (float*)d_out;

    dim3 grid(B_DIM * C_DIM * NPG);   // 192 workgroups, one per (b,c,5 p's)
    dim3 block(NTHREADS);             // 10 wave32 waves = (nTile, p) pairs
    ModulationIndex_11622181503726_kernel<<<grid, block, 0, stream>>>(
        pha, amp, cuts, out);
}
